// Convattn_2765958938649
// MI455X (gfx1250) — compile-verified, matrix operand order checked by reading, not run on hardware
//
#include <hip/hip_runtime.h>
#include <hip/hip_bf16.h>

// ---------------------------------------------------------------------------
// Types for CDNA5 WMMA (gfx1250, wave32)
// ---------------------------------------------------------------------------
typedef __attribute__((ext_vector_type(16))) __bf16 v16bf;
typedef __attribute__((ext_vector_type(8)))  float  v8f;

#define TEMP_N   256
#define TARG_N   484
#define NTOK     740
#define NPAD     768          // padded token count (multiple of 32)
#define DIM      768
#define HEADS    12
#define HD       64
#define BATCH    16
#define POS_HID  512

#if defined(__has_builtin)
#if __has_builtin(__builtin_amdgcn_global_load_async_to_lds_b128) && \
    __has_builtin(__builtin_amdgcn_s_wait_asynccnt)
#define USE_ASYNC_LDS 1
#endif
#endif

// Pointer types expected by the async-to-LDS builtin (from hipcc diagnostics:
// param 1 is "__attribute__((__vector_size__(4 * sizeof(int)))) int __device__ *")
typedef int vi4 __attribute__((vector_size(16)));
typedef __attribute__((address_space(1))) vi4* async_gptr;
typedef __attribute__((address_space(3))) vi4* async_lptr;

union FragU {
    v16bf v;
    uint4 q[2];
};

// Load one 16x32 (bf16) A/B fragment k-chunk from a row-major tile.
// Layout (ISA 7.12.2, 16-bit A 16x32): lanes 0-15 -> m=lane, k in {kb..kb+7, kb+16..kb+23}
//                                      lanes 16-31 -> m=lane-16, k in {kb+8..kb+15, kb+24..kb+31}
static __device__ inline v16bf load_frag(const __bf16* base, int row, int stride,
                                         int kbase, int lane) {
    const __bf16* p = base + (size_t)row * stride + kbase + ((lane & 16) ? 8 : 0);
    FragU u;
    u.q[0] = *(const uint4*)(p);
    u.q[1] = *(const uint4*)(p + 16);
    return u.v;
}

static __device__ inline v8f wmma_bf16(v16bf a, v16bf b, v8f c) {
    return __builtin_amdgcn_wmma_f32_16x16x32_bf16(false, a, false, b,
                                                   (short)0, c, false, false);
}

// ---------------------------------------------------------------------------
// Kernel 1: weight transpose + fp32 -> bf16 convert.
// wT[z][n][k] = w_z[k][n]  (z: 0=q, 1=k, 2=v, 3=proj)
// ---------------------------------------------------------------------------
__global__ void wconv(const float* __restrict__ qw, const float* __restrict__ kw,
                      const float* __restrict__ vw, const float* __restrict__ pw,
                      __bf16* __restrict__ wT) {
    int id = blockIdx.x * 256 + threadIdx.x;
    if (id >= 4 * DIM * DIM) return;
    int z = id / (DIM * DIM);
    int r = id % (DIM * DIM);
    int n = r / DIM;
    int k = r % DIM;
    const float* src = (z == 0) ? qw : (z == 1) ? kw : (z == 2) ? vw : pw;
    wT[id] = (__bf16)src[k * DIM + n];
}

// ---------------------------------------------------------------------------
// Kernel 2: cross-bias MLP.  cross[h][g][t], g<484 target token, t<256 temp token
// ---------------------------------------------------------------------------
__global__ __launch_bounds__(128) void cross_mlp(
        const float* __restrict__ tab,  // [484][256][2]
        const float* __restrict__ w1,   // [2][512]
        const float* __restrict__ b1,   // [512]
        const float* __restrict__ w2,   // [512][12]
        const float* __restrict__ b2,   // [12]
        float* __restrict__ cross) {    // [12][484][256]
    __shared__ float w1s[2 * POS_HID];
    __shared__ float b1s[POS_HID];
    __shared__ float w2s[POS_HID * HEADS];
    int tid = threadIdx.x;
    for (int i = tid; i < 2 * POS_HID; i += 128) w1s[i] = w1[i];
    for (int i = tid; i < POS_HID; i += 128) b1s[i] = b1[i];
    for (int i = tid; i < POS_HID * HEADS; i += 128) w2s[i] = w2[i];
    __syncthreads();

    int id = blockIdx.x * 128 + tid;
    if (id >= TARG_N * TEMP_N) return;
    int g = id / TEMP_N;
    int t = id % TEMP_N;
    float a0 = tab[id * 2 + 0];
    float a1 = tab[id * 2 + 1];
    float acc[HEADS];
#pragma unroll
    for (int hh = 0; hh < HEADS; ++hh) acc[hh] = 0.f;
    for (int j = 0; j < POS_HID; ++j) {
        float hv = fmaxf(a0 * w1s[j] + a1 * w1s[POS_HID + j] + b1s[j], 0.f);
#pragma unroll
        for (int hh = 0; hh < HEADS; ++hh) acc[hh] += hv * w2s[j * HEADS + hh];
    }
#pragma unroll
    for (int hh = 0; hh < HEADS; ++hh)
        cross[((size_t)hh * TARG_N + g) * TEMP_N + t] = acc[hh] + b2[hh];
}

// ---------------------------------------------------------------------------
// Kernel 3: fused QKV projection GEMM (bf16 WMMA, f32 accumulate).
//   M = BATCH*NPAD (padded, zero rows for t>=740), N = 768, K = 768.
//   z = blockIdx.z selects Q / K / V.
//   Q -> Qb[b][h][t][d], K -> Kb[b][h][t][d], V -> Vt[b][h][d][t]
//   A tile: fp32 -> bf16 converted through LDS.
//   B tile: bf16 weights staged to LDS with GLOBAL_LOAD_ASYNC_TO_LDS_B128.
// ---------------------------------------------------------------------------
__global__ __launch_bounds__(128) void qkv_gemm(
        const float* __restrict__ x,        // [B][740][768]
        const float* __restrict__ temp_q,   // [B][256][768]
        const float* __restrict__ target_q, // [B][484][768]
        const __bf16* __restrict__ wT,      // [4][768][768] (n-major)
        __bf16* __restrict__ Qb, __bf16* __restrict__ Kb, __bf16* __restrict__ Vt) {
    int z = blockIdx.z;
    const __bf16* w = wT + (size_t)z * DIM * DIM;
    __shared__ __bf16 As[64 * 32];
    __shared__ __bf16 Bs[64 * 32];

    int tid = threadIdx.x;
    int lane = tid & 31;
    int wid = tid >> 5;
    int mBase = blockIdx.x * 64;
    int nBase = blockIdx.y * 64;
    int mSub = (wid >> 1) * 32;
    int nSub = (wid & 1) * 32;

    v8f c[2][2] = {};

    // A staging assignment: 64 rows x 2 half-rows of 16 elements
    int srow = tid >> 1;
    int half = tid & 1;
    int m = mBase + srow;
    int b = m / NPAD;
    int t = m % NPAD;
    const float* src = nullptr;
    if (z == 0) {
        if (t < TEMP_N)      src = temp_q   + ((size_t)b * TEMP_N + t) * DIM;
        else if (t < NTOK)   src = target_q + ((size_t)b * TARG_N + (t - TEMP_N)) * DIM;
    } else {
        if (t < NTOK)        src = x + ((size_t)b * NTOK + t) * DIM;
    }

    // B staging assignment: 256 chunks of 8 bf16 (16 bytes); 2 chunks per thread.
    // chunk c: row = c >> 2 (n within tile), col = (c & 3) * 8 (k within chunk)
    int bch = tid * 2;
    int brow0 = bch >> 2,       bcol0 = (bch & 3) * 8;
    int brow1 = (bch + 1) >> 2, bcol1 = ((bch + 1) & 3) * 8;

    for (int kc = 0; kc < DIM / 32; ++kc) {
        __syncthreads();
        // ---- B tile: global bf16 -> LDS ----
        {
            const __bf16* g0 = w + (size_t)(nBase + brow0) * DIM + kc * 32 + bcol0;
            const __bf16* g1 = w + (size_t)(nBase + brow1) * DIM + kc * 32 + bcol1;
            __bf16* l0 = Bs + brow0 * 32 + bcol0;
            __bf16* l1 = Bs + brow1 * 32 + bcol1;
#ifdef USE_ASYNC_LDS
            __builtin_amdgcn_global_load_async_to_lds_b128(
                (async_gptr)g0, (async_lptr)l0, 0, 0);
            __builtin_amdgcn_global_load_async_to_lds_b128(
                (async_gptr)g1, (async_lptr)l1, 0, 0);
#else
            *(uint4*)l0 = *(const uint4*)g0;
            *(uint4*)l1 = *(const uint4*)g1;
#endif
        }
        // ---- A tile: global fp32 -> bf16 -> LDS ----
        {
            int colb = half * 16;
            __bf16* d = As + srow * 32 + colb;
            if (src) {
                const float4* s4 = (const float4*)(src + kc * 32 + colb);
                float4 f0 = s4[0], f1 = s4[1], f2 = s4[2], f3 = s4[3];
                d[0]  = (__bf16)f0.x; d[1]  = (__bf16)f0.y; d[2]  = (__bf16)f0.z; d[3]  = (__bf16)f0.w;
                d[4]  = (__bf16)f1.x; d[5]  = (__bf16)f1.y; d[6]  = (__bf16)f1.z; d[7]  = (__bf16)f1.w;
                d[8]  = (__bf16)f2.x; d[9]  = (__bf16)f2.y; d[10] = (__bf16)f2.z; d[11] = (__bf16)f2.w;
                d[12] = (__bf16)f3.x; d[13] = (__bf16)f3.y; d[14] = (__bf16)f3.z; d[15] = (__bf16)f3.w;
            } else {
#pragma unroll
                for (int j = 0; j < 16; ++j) d[j] = (__bf16)0.0f;
            }
        }
#ifdef USE_ASYNC_LDS
        __builtin_amdgcn_s_wait_asynccnt(0);
#endif
        __syncthreads();

        v16bf a0 = load_frag(As, mSub + (lane & 15),      32, 0, lane);
        v16bf a1 = load_frag(As, mSub + 16 + (lane & 15), 32, 0, lane);
        v16bf b0 = load_frag(Bs, nSub + (lane & 15),      32, 0, lane);
        v16bf b1 = load_frag(Bs, nSub + 16 + (lane & 15), 32, 0, lane);
        c[0][0] = wmma_bf16(a0, b0, c[0][0]);
        c[0][1] = wmma_bf16(a0, b1, c[0][1]);
        c[1][0] = wmma_bf16(a1, b0, c[1][0]);
        c[1][1] = wmma_bf16(a1, b1, c[1][1]);
    }

    int rloc = (lane & 16) ? 8 : 0;
    int col = lane & 15;
#pragma unroll
    for (int im = 0; im < 2; ++im)
#pragma unroll
        for (int in = 0; in < 2; ++in)
#pragma unroll
            for (int i = 0; i < 8; ++i) {
                int mm = mBase + mSub + 16 * im + i + rloc;
                int nn = nBase + nSub + 16 * in + col;
                int bb = mm / NPAD, tt = mm % NPAD;
                int h = nn >> 6, dd = nn & 63;
                float v = c[im][in][i];
                if (z == 0)
                    Qb[(((size_t)bb * HEADS + h) * NPAD + tt) * HD + dd] = (__bf16)v;
                else if (z == 1)
                    Kb[(((size_t)bb * HEADS + h) * NPAD + tt) * HD + dd] = (__bf16)v;
                else
                    Vt[(((size_t)bb * HEADS + h) * HD + dd) * NPAD + tt] = (__bf16)v;
            }
}

// ---------------------------------------------------------------------------
// Bias gather (computed on the fly; no materialized [H][N][N] matrix)
// ---------------------------------------------------------------------------
static __device__ inline float bias_at(int h, int q, int k,
                                       const float* __restrict__ rpb,
                                       const float* __restrict__ rpb_t,
                                       const float* __restrict__ cross,
                                       const int* __restrict__ tidx,
                                       const int* __restrict__ gidx) {
    if (k >= NTOK || q >= NTOK) return -1e9f;
    if (q < TEMP_N) {
        if (k < TEMP_N) return rpb_t[tidx[q * TEMP_N + k] * HEADS + h];
        return cross[((size_t)h * TARG_N + (k - TEMP_N)) * TEMP_N + q];
    }
    int qg = q - TEMP_N;
    if (k < TEMP_N) return cross[((size_t)h * TARG_N + qg) * TEMP_N + k];
    return rpb[gidx[qg * TARG_N + (k - TEMP_N)] * HEADS + h];
}

// ---------------------------------------------------------------------------
// Kernel 4: flash-style attention, one wave per (b, h, 16-query tile).
// ---------------------------------------------------------------------------
__global__ __launch_bounds__(32) void attn(
        const __bf16* __restrict__ Qb, const __bf16* __restrict__ Kb,
        const __bf16* __restrict__ Vt,
        const float* __restrict__ rpb, const float* __restrict__ rpb_t,
        const float* __restrict__ cross,
        const int* __restrict__ tidx, const int* __restrict__ gidx,
        __bf16* __restrict__ ctx) {
    int lane = threadIdx.x;
    int qt = blockIdx.x, h = blockIdx.y, b = blockIdx.z;
    int qBase = qt * 16;
    int col = lane & 15;
    int rloc = (lane & 16) ? 8 : 0;

    const __bf16* Qp = Qb + ((size_t)b * HEADS + h) * NPAD * HD;
    const __bf16* Kp = Kb + ((size_t)b * HEADS + h) * NPAD * HD;
    const __bf16* Vp = Vt + ((size_t)b * HEADS + h) * HD * NPAD;

    v16bf aq0 = load_frag(Qp, qBase + col, HD, 0, lane);
    v16bf aq1 = load_frag(Qp, qBase + col, HD, 32, lane);

    v8f o[4] = {};
    float mrow[8], lrow[8];
#pragma unroll
    for (int i = 0; i < 8; ++i) { mrow[i] = -1e30f; lrow[i] = 0.f; }

    __shared__ __bf16 P[16 * 32];
    const float scale = 0.125f; // hd^-0.5 = 1/8

    for (int kt = 0; kt < NPAD; kt += 32) {
        v8f s0 = {}, s1 = {};
        {
            v16bf bk;
            bk = load_frag(Kp, kt + col,      HD, 0,  lane); s0 = wmma_bf16(aq0, bk, s0);
            bk = load_frag(Kp, kt + col,      HD, 32, lane); s0 = wmma_bf16(aq1, bk, s0);
            bk = load_frag(Kp, kt + 16 + col, HD, 0,  lane); s1 = wmma_bf16(aq0, bk, s1);
            bk = load_frag(Kp, kt + 16 + col, HD, 32, lane); s1 = wmma_bf16(aq1, bk, s1);
        }
#pragma unroll
        for (int i = 0; i < 8; ++i) {
            int q = qBase + i + rloc;
            float sv0 = s0[i] * scale + bias_at(h, q, kt + col,      rpb, rpb_t, cross, tidx, gidx);
            float sv1 = s1[i] * scale + bias_at(h, q, kt + 16 + col, rpb, rpb_t, cross, tidx, gidx);
            // row max across the 16 lanes of this half-wave
            float v = fmaxf(sv0, sv1);
            v = fmaxf(v, __shfl_xor(v, 8, 32));
            v = fmaxf(v, __shfl_xor(v, 4, 32));
            v = fmaxf(v, __shfl_xor(v, 2, 32));
            v = fmaxf(v, __shfl_xor(v, 1, 32));
            float m_new = fmaxf(mrow[i], v);
            float f = __expf(mrow[i] - m_new);
            mrow[i] = m_new;
            float p0 = __expf(sv0 - m_new);
            float p1 = __expf(sv1 - m_new);
            float ps = p0 + p1;
            ps += __shfl_xor(ps, 8, 32);
            ps += __shfl_xor(ps, 4, 32);
            ps += __shfl_xor(ps, 2, 32);
            ps += __shfl_xor(ps, 1, 32);
            lrow[i] = lrow[i] * f + ps;
#pragma unroll
            for (int j = 0; j < 4; ++j) o[j][i] *= f;
            P[(i + rloc) * 32 + col]      = (__bf16)p0;
            P[(i + rloc) * 32 + 16 + col] = (__bf16)p1;
        }
        __syncthreads();
        v16bf ap = load_frag(P, col, 32, 0, lane);
#pragma unroll
        for (int j = 0; j < 4; ++j) {
            v16bf bv = load_frag(Vp, 16 * j + col, NPAD, kt, lane);
            o[j] = wmma_bf16(ap, bv, o[j]);
        }
        __syncthreads();
    }

#pragma unroll
    for (int i = 0; i < 8; ++i) {
        float inv = 1.0f / lrow[i];
        int t = qBase + i + rloc;
#pragma unroll
        for (int j = 0; j < 4; ++j)
            ctx[((size_t)(b * NPAD + t)) * DIM + h * HD + 16 * j + col] =
                (__bf16)(o[j][i] * inv);
    }
}

// ---------------------------------------------------------------------------
// Kernel 5: output projection GEMM (direct global bf16 fragments) + bias.
// ---------------------------------------------------------------------------
__global__ __launch_bounds__(128) void proj_gemm(
        const __bf16* __restrict__ ctx,   // [B*NPAD][768] bf16
        const __bf16* __restrict__ wTp,   // proj wT [768][768] (n-major)
        const float* __restrict__ proj_b, // [768]
        float* __restrict__ out) {        // [B][740][768]
    int tid = threadIdx.x;
    int lane = tid & 31;
    int wid = tid >> 5;
    int mBase = blockIdx.x * 64 + (wid >> 1) * 32;
    int nBase = blockIdx.y * 64 + (wid & 1) * 32;
    int col = lane & 15;
    int rloc = (lane & 16) ? 8 : 0;

    v8f c[2][2] = {};
    for (int kc = 0; kc < DIM / 32; ++kc) {
        v16bf a0 = load_frag(ctx, mBase + col,      DIM, kc * 32, lane);
        v16bf a1 = load_frag(ctx, mBase + 16 + col, DIM, kc * 32, lane);
        v16bf b0 = load_frag(wTp, nBase + col,      DIM, kc * 32, lane);
        v16bf b1 = load_frag(wTp, nBase + 16 + col, DIM, kc * 32, lane);
        c[0][0] = wmma_bf16(a0, b0, c[0][0]);
        c[0][1] = wmma_bf16(a0, b1, c[0][1]);
        c[1][0] = wmma_bf16(a1, b0, c[1][0]);
        c[1][1] = wmma_bf16(a1, b1, c[1][1]);
    }

#pragma unroll
    for (int in = 0; in < 2; ++in) {
        int nn = nBase + 16 * in + col;
        float pb = proj_b[nn];
#pragma unroll
        for (int im = 0; im < 2; ++im)
#pragma unroll
            for (int i = 0; i < 8; ++i) {
                int mm = mBase + 16 * im + i + rloc;
                int bb = mm / NPAD, tt = mm % NPAD;
                if (tt < NTOK)
                    out[((size_t)bb * NTOK + tt) * DIM + nn] = c[im][in][i] + pb;
            }
    }
}

// ---------------------------------------------------------------------------
// Host-side launcher
// ---------------------------------------------------------------------------
extern "C" void kernel_launch(void* const* d_in, const int* in_sizes, int n_in,
                              void* d_out, int out_size, void* d_ws, size_t ws_size,
                              hipStream_t stream) {
    const float* x        = (const float*)d_in[0];
    const float* temp_q   = (const float*)d_in[1];
    const float* target_q = (const float*)d_in[2];
    const float* q_w      = (const float*)d_in[3];
    const float* k_w      = (const float*)d_in[4];
    const float* v_w      = (const float*)d_in[5];
    const float* proj_w   = (const float*)d_in[6];
    const float* proj_b   = (const float*)d_in[7];
    const float* rpb      = (const float*)d_in[8];   // [(2*22-1)^2][12]
    const float* rpb_t    = (const float*)d_in[9];   // [(2*16-1)^2][12]
    const float* pos_w1   = (const float*)d_in[10];
    const float* pos_b1   = (const float*)d_in[11];
    const float* pos_w2   = (const float*)d_in[12];
    const float* pos_b2   = (const float*)d_in[13];
    const float* tab      = (const float*)d_in[14];  // [484][256][2]
    const int*   tidx     = (const int*)d_in[15];    // [256][256]
    const int*   gidx     = (const int*)d_in[16];    // [484][484]
    float* out = (float*)d_out;

    // workspace layout (bytes, 256-aligned)
    char* ws = (char*)d_ws;
    const size_t OFF_WT    = 0;                               // 4*768*768*2
    const size_t OFF_QB    = OFF_WT + (size_t)4 * DIM * DIM * 2;
    const size_t SZ_QKV    = (size_t)BATCH * HEADS * NPAD * HD * 2;
    const size_t OFF_KB    = OFF_QB + SZ_QKV;
    const size_t OFF_VT    = OFF_KB + SZ_QKV;
    const size_t OFF_CROSS = OFF_VT + SZ_QKV;                 // 12*484*256*4
    const size_t OFF_CTX   = OFF_CROSS + (size_t)HEADS * TARG_N * TEMP_N * 4;

    __bf16* wT    = (__bf16*)(ws + OFF_WT);
    __bf16* Qb    = (__bf16*)(ws + OFF_QB);
    __bf16* Kb    = (__bf16*)(ws + OFF_KB);
    __bf16* Vt    = (__bf16*)(ws + OFF_VT);
    float*  cross = (float*)(ws + OFF_CROSS);
    __bf16* ctx   = (__bf16*)(ws + OFF_CTX);

    // 1. weight transpose + convert
    wconv<<<dim3((4 * DIM * DIM + 255) / 256), dim3(256), 0, stream>>>(
        q_w, k_w, v_w, proj_w, wT);

    // 2. cross-bias MLP
    cross_mlp<<<dim3((TARG_N * TEMP_N + 127) / 128), dim3(128), 0, stream>>>(
        tab, pos_w1, pos_b1, pos_w2, pos_b2, cross);

    // 3. fused QKV projection
    qkv_gemm<<<dim3(BATCH * NPAD / 64, DIM / 64, 3), dim3(128), 0, stream>>>(
        x, temp_q, target_q, wT, Qb, Kb, Vt);

    // 4. attention
    attn<<<dim3(NPAD / 16, HEADS, BATCH), dim3(32), 0, stream>>>(
        Qb, Kb, Vt, rpb, rpb_t, cross, tidx, gidx, ctx);

    // 5. output projection
    proj_gemm<<<dim3(BATCH * NPAD / 64, DIM / 64), dim3(128), 0, stream>>>(
        ctx, wT + (size_t)3 * DIM * DIM, proj_b, out);
}
